// CTMCore_31722628448771
// MI455X (gfx1250) — matrix operand order repaired and checked by reading, not compiled, not run on hardware
//
#include <hip/hip_runtime.h>
#include <hip/hip_bf16.h>
#include <math.h>

#define B_   256
#define L_   4096
#define E_   64
#define D_   128
#define H_   4
#define DH_  16
#define SH_  256
#define M_   8
#define NED_ 16
#define NH_  64
#define T_   8

#define ROWS_PER_STAGE 128
#define N_STAGES (L_ / ROWS_PER_STAGE)   // 32
#define LDS_ROW 68                        // 64 + 4 pad dwords (TDM pad feature)

typedef __attribute__((ext_vector_type(16))) __bf16 v16bf;
typedef __attribute__((ext_vector_type(8)))  float  v8f;
typedef __attribute__((ext_vector_type(2)))  float  v2f;
typedef unsigned int u32x4 __attribute__((ext_vector_type(4)));
typedef int          i32x8 __attribute__((ext_vector_type(8)));
typedef int          i32x4 __attribute__((ext_vector_type(4)));

// ---------------------------------------------------------------------------
// init: broadcast z0 -> z (B,D), a0 -> ah (B,D,M)
// ---------------------------------------------------------------------------
__global__ __launch_bounds__(256) void init_state(const float* __restrict__ z0,
                                                  const float* __restrict__ a0,
                                                  float* __restrict__ z,
                                                  float* __restrict__ ah) {
  int i = blockIdx.x * 256 + threadIdx.x;   // over B*D
  int d = i & (D_ - 1);
  z[i] = z0[d];
  for (int mm = 0; mm < M_; ++mm) ah[(size_t)i * M_ + mm] = a0[d * M_ + mm];
}

// ---------------------------------------------------------------------------
// query head: LayerNorm(z) -> q = zn@Wq.T+bq -> qh = q@Wiq.T+biq   (per b)
// ---------------------------------------------------------------------------
__global__ __launch_bounds__(128) void qhead_kernel(
    const float* __restrict__ z, const float* __restrict__ ln_g,
    const float* __restrict__ ln_b, const float* __restrict__ Wq,
    const float* __restrict__ bq, const float* __restrict__ Wiq,
    const float* __restrict__ biq, float* __restrict__ qhout) {
  const int b = blockIdx.x, d = threadIdx.x;
  __shared__ float red[D_], znS[D_], qS[E_];
  float zv = z[b * D_ + d];
  red[d] = zv; __syncthreads();
  for (int s = 64; s > 0; s >>= 1) { if (d < s) red[d] += red[d + s]; __syncthreads(); }
  float mu = red[0] * (1.f / D_);
  __syncthreads();
  float df = zv - mu;
  red[d] = df * df; __syncthreads();
  for (int s = 64; s > 0; s >>= 1) { if (d < s) red[d] += red[d + s]; __syncthreads(); }
  float var = red[0] * (1.f / D_);
  znS[d] = df * rsqrtf(var + 1e-5f) * ln_g[d] + ln_b[d];
  __syncthreads();
  if (d < E_) {
    float acc = bq[d];
    const float* wr = Wq + d * D_;
    for (int k = 0; k < D_; ++k) acc += znS[k] * wr[k];
    qS[d] = acc;
  }
  __syncthreads();
  if (d < E_) {
    float acc = biq[d];
    const float* wr = Wiq + d * E_;
    for (int e = 0; e < E_; ++e) acc += qS[e] * wr[e];
    qhout[b * E_ + d] = acc;
  }
}

// ---------------------------------------------------------------------------
// TDM: DMA one 128x64-dword token slab into LDS, padded to 68-dword rows so
// consecutive rows land in distinct LDS banks for the WMMA A-tile reads.
// D# built per CDNA5 ISA ch.8 (group0: count/lds/global/type, group1: dims).
// This toolchain exposes the 6-arg builtin: (g0,g1,g2,g3,ext,cpol).
// ---------------------------------------------------------------------------
__device__ __forceinline__ void tdm_load_slab(const float* gsrc, unsigned lds_off) {
  unsigned long long ga = (unsigned long long)(size_t)gsrc;
  u32x4 g0;
  g0[0] = 1u;                                        // count=1, user descriptor
  g0[1] = lds_off;                                   // lds_addr (bytes)
  g0[2] = (unsigned)(ga & 0xFFFFFFFFu);              // global_addr[31:0]
  g0[3] = (unsigned)((ga >> 32) & 0x01FFFFFFu)       // global_addr[56:32]
        | (2u << 30);                                // type = 2 ("image")
  i32x8 g1;
  g1[0] = (2 << 16)        // data_size = 4B
        | (1 << 20)        // pad_enable
        | (5 << 22)        // pad_interval: 2^(5+1)=64 dwords
        | (3 << 25);       // pad_amount: 3+1 = 4 dwords
  g1[1] = (E_ << 16);                  // tensor_dim0 = 64 (low 16 bits)
  g1[2] = (int)((unsigned)L_ << 16);   // tensor_dim1 = 4096 (low 16 bits)
  g1[3] = (E_ << 16);                  // tile_dim0 = 64
  g1[4] = ROWS_PER_STAGE;              // tile_dim1 = 128
  g1[5] = E_;                          // tensor_dim0_stride = 64
  g1[6] = 0;
  g1[7] = 0;
  i32x4 gz4 = {0, 0, 0, 0};            // groups 2/3 unused (2D tensor)
  i32x8 gz8 = {0, 0, 0, 0, 0, 0, 0, 0};
  __builtin_amdgcn_tensor_load_to_lds(g0, g1, gz4, gz4, gz8, 0);
}

// ---------------------------------------------------------------------------
// Fused K/V projection + single-query flash attention.
// One block per batch b; 8 waves = 4 heads x 2 row-interleaves. Token slabs
// are DMA'd into LDS once by the TDM (double-buffered) and shared by all
// heads; each wave projects 16-row chunks to K/V tiles via bf16 WMMA and
// maintains an online softmax.
// ---------------------------------------------------------------------------
__global__ __launch_bounds__(256) void attn_fused(
    const float* __restrict__ tokens, const float* __restrict__ Wik,
    const float* __restrict__ bik, const float* __restrict__ Wiv,
    const float* __restrict__ biv, const float* __restrict__ qh,
    float* __restrict__ o_out) {
  const int b     = blockIdx.x;
  const int tid   = threadIdx.x;
  const int wid   = tid >> 5;
  const int lane  = tid & 31;
  const int h     = wid >> 1;
  const int halfw = wid & 1;
  const int row16 = lane & 15;
  const int hi    = lane >> 4;

  __shared__ float tok_s[2][ROWS_PER_STAGE][LDS_ROW];   // ~68 KB, double buffer
  __shared__ float sm[8], ssh[8], soh[8][16];

  // Weight B-tiles: k = tok @ Wik.T  =>  B[k][n] = Wik[h*16+n][k]
  // 16-bit B 32x16 as loaded here: lane = col n, halves hold K 0-15 / 16-31 (same order as A).
  v16bf Bk[2], Bv[2];
  {
    const int wr = h * DH_ + row16;
    for (int ks = 0; ks < 2; ++ks) {
      const float* pk = Wik + wr * E_ + ks * 32 + hi * 16;
      const float* pv = Wiv + wr * E_ + ks * 32 + hi * 16;
      for (int e = 0; e < 16; ++e) {
        Bk[ks][e] = (__bf16)pk[e];
        Bv[ks][e] = (__bf16)pv[e];
      }
    }
  }
  const float qv = qh[(b * H_ + h) * DH_ + row16];
  const float bk = bik[h * DH_ + row16];
  const float bv = biv[h * DH_ + row16];
  const float scale = 0.25f;   // 1/sqrt(DH)

  float m = -1e30f, ssum = 0.f, oacc = 0.f;
  const float* tokb = tokens + (size_t)b * L_ * E_;

  // prologue: DMA stages 0 and 1
  if (wid == 0) {
    tdm_load_slab(tokb, (unsigned)(size_t)&tok_s[0][0][0]);
    tdm_load_slab(tokb + (size_t)ROWS_PER_STAGE * E_,
                  (unsigned)(size_t)&tok_s[1][0][0]);
  }

  for (int s = 0; s < N_STAGES; ++s) {
    if (wid == 0) {
      if (s == N_STAGES - 1) __builtin_amdgcn_s_wait_tensorcnt(0);
      else                   __builtin_amdgcn_s_wait_tensorcnt(1);
    }
    __syncthreads();   // slab s visible to all waves

    const int buf = s & 1;
    for (int j = 0; j < 4; ++j) {
      const int sub  = j * 2 + halfw;                  // this wave's 16-row chunk
      const float* prow = &tok_s[buf][sub * 16 + row16][0];

      // A tiles (16-bit A 16x32): lane = row, element e -> K=hi*16+e
      v16bf a0, a1;
      {
        const float* p0 = prow + hi * 16;        // k-step 0
        const float* p1 = prow + 32 + hi * 16;   // k-step 1
        for (int e = 0; e < 8; ++e) {
          a0[e]     = (__bf16)p0[e];
          a0[e + 8] = (__bf16)p0[e + 8];
          a1[e]     = (__bf16)p1[e];
          a1[e + 8] = (__bf16)p1[e + 8];
        }
      }
      v8f kacc = {}; v8f vacc = {};
      kacc = __builtin_amdgcn_wmma_f32_16x16x32_bf16(false, a0, false, Bk[0], (short)0, kacc, false, false);
      kacc = __builtin_amdgcn_wmma_f32_16x16x32_bf16(false, a1, false, Bk[1], (short)0, kacc, false, false);
      vacc = __builtin_amdgcn_wmma_f32_16x16x32_bf16(false, a0, false, Bv[0], (short)0, vacc, false, false);
      vacc = __builtin_amdgcn_wmma_f32_16x16x32_bf16(false, a1, false, Bv[1], (short)0, vacc, false, false);

      // scores: reduce over dh (the 16-lane groups) with wave32 butterflies
      float sr[8];
      float cmax = -1e30f;
      for (int r = 0; r < 8; ++r) {
        float sv = (kacc[r] + bk) * qv;
        sv += __shfl_xor(sv, 1);
        sv += __shfl_xor(sv, 2);
        sv += __shfl_xor(sv, 4);
        sv += __shfl_xor(sv, 8);
        sv *= scale;
        sr[r] = sv;
        cmax = fmaxf(cmax, sv);
      }
      float mnew = fmaxf(m, cmax);
      float f = __expf(m - mnew);
      ssum *= f; oacc *= f;
      for (int r = 0; r < 8; ++r) {
        float p = __expf(sr[r] - mnew);
        ssum += p;
        oacc += p * (vacc[r] + bv);   // lane's dh column, its half's row r
      }
      m = mnew;
    }

    __syncthreads();   // everyone done reading buf before it is re-DMAed
    if (wid == 0 && s + 2 < N_STAGES) {
      tdm_load_slab(tokb + (size_t)(s + 2) * ROWS_PER_STAGE * E_,
                    (unsigned)(size_t)&tok_s[buf][0][0]);
    }
  }

  // combine lane halves (rows 0-7 group vs 8-15 group) — same dh at lane^16
  float m2 = __shfl_xor(m, 16);
  float s2 = __shfl_xor(ssum, 16);
  float o2 = __shfl_xor(oacc, 16);
  float mf = fmaxf(m, m2);
  float e1 = __expf(m - mf), e2 = __expf(m2 - mf);
  float sf = ssum * e1 + s2 * e2;
  float of = oacc * e1 + o2 * e2;

  // combine the two waves of each head via LDS
  if (lane == 0) { sm[wid] = mf; ssh[wid] = sf; }
  if (lane < 16) soh[wid][lane] = of;
  __syncthreads();
  if (tid < H_ * DH_) {
    const int hh = tid >> 4, dh = tid & 15;
    const int w0 = hh * 2, w1 = w0 + 1;
    float mA = sm[w0], mB = sm[w1];
    float mm2 = fmaxf(mA, mB);
    float fA = __expf(mA - mm2), fB = __expf(mB - mm2);
    float den = ssh[w0] * fA + ssh[w1] * fB;
    float num = soh[w0][dh] * fA + soh[w1][dh] * fB;
    o_out[(b * H_ + hh) * DH_ + dh] = num / den;
  }
}

// ---------------------------------------------------------------------------
// x = concat(z, o@Wo.T + bo)   (B x 192)
// ---------------------------------------------------------------------------
__global__ __launch_bounds__(256) void oproj_concat(
    const float* __restrict__ z, const float* __restrict__ o,
    const float* __restrict__ Wo, const float* __restrict__ bo,
    float* __restrict__ x) {
  int i = blockIdx.x * 256 + threadIdx.x;   // B*(D+E)
  int b = i / (D_ + E_), c = i % (D_ + E_);
  if (c < D_) {
    x[i] = z[b * D_ + c];
  } else {
    int e = c - D_;
    float acc = bo[e];
    const float* ob = o + b * E_;
    const float* wr = Wo + e * E_;
    for (int j = 0; j < E_; ++j) acc += ob[j] * wr[j];
    x[i] = acc;
  }
}

// ---------------------------------------------------------------------------
// Full-precision GEMM via V_WMMA_F32_16X16X4_F32:
// Out[row][col] = bias[col] + sum_k A[row][k] * Wt[col][k]
// grid = (Ncols/16, Mrows/16), one wave per 16x16 tile.
// f32 A 16x4: lane = row, halves hold K pair {0,1}/{2,3}. B mirrored.
// ---------------------------------------------------------------------------
__global__ __launch_bounds__(32) void gemm_f32_wmma(
    const float* __restrict__ A, const float* __restrict__ Wt,
    const float* __restrict__ bias, float* __restrict__ Out,
    int K, int Ncols) {
  const int ct = blockIdx.x, rt = blockIdx.y;
  const int lane = threadIdx.x;
  const int row16 = lane & 15, hi = lane >> 4;
  const int arow = rt * 16 + row16;
  const int wrow = ct * 16 + row16;
  v8f acc = {};
  for (int k0 = 0; k0 < K; k0 += 4) {
    const int kb = k0 + hi * 2;
    v2f av, bw;
    av.x = A[arow * K + kb];
    av.y = A[arow * K + kb + 1];
    bw.x = Wt[wrow * K + kb];
    bw.y = Wt[wrow * K + kb + 1];
    acc = __builtin_amdgcn_wmma_f32_16x16x4_f32(false, av, false, bw, (short)0, acc, false, false);
  }
  const float bb = bias[ct * 16 + row16];
  for (int r = 0; r < 8; ++r) {
    const int orow = rt * 16 + r + hi * 8;
    Out[orow * Ncols + ct * 16 + row16] = acc[r] + bb;
  }
}

// ---------------------------------------------------------------------------
// GLU gate:  ap = g1 * sigmoid(g2)
// ---------------------------------------------------------------------------
__global__ __launch_bounds__(256) void glu_kernel(const float* __restrict__ s,
                                                  float* __restrict__ ap) {
  int i = blockIdx.x * 256 + threadIdx.x;   // B*SH
  int b = i >> 8, j = i & (SH_ - 1);
  float g1 = s[b * 2 * SH_ + j];
  float g2 = s[b * 2 * SH_ + SH_ + j];
  ap[i] = g1 * (1.f / (1.f + __expf(-g2)));
}

// ---------------------------------------------------------------------------
// roll history + per-neuron NLM (shared weights in LDS) + tanh + emit z_t
// ---------------------------------------------------------------------------
__global__ __launch_bounds__(256) void nlm_kernel(
    const float* __restrict__ a, const float* __restrict__ Wn1,
    const float* __restrict__ bn1, const float* __restrict__ Wn2,
    const float* __restrict__ bn2, const float* __restrict__ emb,
    float* __restrict__ ah, float* __restrict__ z,
    float* __restrict__ out, int tick) {
  __shared__ float W1s[NH_ * (M_ + NED_)], b1s[NH_], W2s[NH_];
  for (int i = threadIdx.x; i < NH_ * (M_ + NED_); i += 256) W1s[i] = Wn1[i];
  if (threadIdx.x < NH_) {
    b1s[threadIdx.x] = bn1[threadIdx.x];
    W2s[threadIdx.x] = Wn2[threadIdx.x];
  }
  __syncthreads();
  int i = blockIdx.x * 256 + threadIdx.x;   // B*D
  int b = i >> 7, d = i & (D_ - 1);
  float hv[M_];
  float* ahp = ah + (size_t)i * M_;
  for (int mm = 0; mm < M_ - 1; ++mm) hv[mm] = ahp[mm + 1];
  hv[M_ - 1] = a[i];
  for (int mm = 0; mm < M_; ++mm) ahp[mm] = hv[mm];
  float ev[NED_];
  const float* ep = emb + d * NED_;
  for (int e = 0; e < NED_; ++e) ev[e] = ep[e];
  float acc2 = bn2[0];
  for (int j = 0; j < NH_; ++j) {
    float t = b1s[j];
    const float* wr = &W1s[j * (M_ + NED_)];
    for (int mm = 0; mm < M_; ++mm) t += hv[mm] * wr[mm];
    for (int e = 0; e < NED_; ++e) t += ev[e] * wr[M_ + e];
    float sj = t * (1.f / (1.f + __expf(-t)));   // silu
    acc2 += sj * W2s[j];
  }
  float zv = tanhf(acc2);
  z[i] = zv;
  out[((size_t)b * T_ + tick) * D_ + d] = zv;
}

// ---------------------------------------------------------------------------
extern "C" void kernel_launch(void* const* d_in, const int* in_sizes, int n_in,
                              void* d_out, int out_size, void* d_ws, size_t ws_size,
                              hipStream_t stream) {
  (void)in_sizes; (void)n_in; (void)out_size; (void)ws_size;
  const float* tokens = (const float*)d_in[0];
  const float* z0   = (const float*)d_in[1];
  const float* a0   = (const float*)d_in[2];
  const float* ln_g = (const float*)d_in[3];
  const float* ln_b = (const float*)d_in[4];
  const float* Wq   = (const float*)d_in[5];
  const float* bq   = (const float*)d_in[6];
  const float* Wiq  = (const float*)d_in[7];
  const float* biq  = (const float*)d_in[8];
  const float* Wik  = (const float*)d_in[9];
  const float* bik  = (const float*)d_in[10];
  const float* Wiv  = (const float*)d_in[11];
  const float* biv  = (const float*)d_in[12];
  const float* Wo   = (const float*)d_in[13];
  const float* bo   = (const float*)d_in[14];
  const float* Ws1  = (const float*)d_in[15];
  const float* bs1  = (const float*)d_in[16];
  const float* Ws2  = (const float*)d_in[17];
  const float* bs2  = (const float*)d_in[18];
  const float* Wn1  = (const float*)d_in[19];
  const float* bn1  = (const float*)d_in[20];
  const float* Wn2  = (const float*)d_in[21];
  const float* bn2  = (const float*)d_in[22];
  const float* emb  = (const float*)d_in[23];
  float* out = (float*)d_out;

  float* ws  = (float*)d_ws;
  float* z   = ws;                    // B*D
  float* ah  = z   + B_ * D_;         // B*D*M
  float* qhb = ah  + B_ * D_ * M_;    // B*E
  float* ob  = qhb + B_ * E_;         // B*E
  float* xb  = ob  + B_ * E_;         // B*(D+E)
  float* sb  = xb  + B_ * (D_ + E_);  // B*2SH
  float* apb = sb  + B_ * 2 * SH_;    // B*SH
  float* ab  = apb + B_ * SH_;        // B*D

  init_state<<<(B_ * D_) / 256, 256, 0, stream>>>(z0, a0, z, ah);
  for (int t = 0; t < T_; ++t) {
    qhead_kernel<<<B_, 128, 0, stream>>>(z, ln_g, ln_b, Wq, bq, Wiq, biq, qhb);
    attn_fused<<<B_, 256, 0, stream>>>(tokens, Wik, bik, Wiv, biv, qhb, ob);
    oproj_concat<<<(B_ * (D_ + E_)) / 256, 256, 0, stream>>>(z, ob, Wo, bo, xb);
    gemm_f32_wmma<<<dim3(2 * SH_ / 16, B_ / 16), 32, 0, stream>>>(
        xb, Ws1, bs1, sb, D_ + E_, 2 * SH_);
    glu_kernel<<<(B_ * SH_) / 256, 256, 0, stream>>>(sb, apb);
    gemm_f32_wmma<<<dim3(D_ / 16, B_ / 16), 32, 0, stream>>>(
        apb, Ws2, bs2, ab, SH_, D_);
    nlm_kernel<<<(B_ * D_) / 256, 256, 0, stream>>>(
        ab, Wn1, bn1, Wn2, bn2, emb, ah, z, out, t);
  }
}
